// RelativeMultiHeadSelfAttention_3573412790778
// MI455X (gfx1250) — matrix operand order repaired and checked by reading, not compiled, not run on hardware
//
#include <hip/hip_runtime.h>

#define SEQ   2048
#define DIM   1024
#define NH    16
#define HD    64
#define BATCH 2

typedef __attribute__((ext_vector_type(16))) __bf16 bf16x16;
typedef __attribute__((ext_vector_type(8)))  __bf16 bf16x8;
typedef __attribute__((ext_vector_type(8)))  float  f32x8;

__device__ __forceinline__ unsigned short f2bf(float f) {
  unsigned int u = __float_as_uint(f);
  u += 0x7FFFu + ((u >> 16) & 1u);           // round-to-nearest-even
  return (unsigned short)(u >> 16);
}
__device__ __forceinline__ float bf2f(unsigned short h) {
  return __uint_as_float(((unsigned int)h) << 16);
}

// A-matrix 16x32 bf16 fragment: lane holds row m=lane&15,
// elems = A[m][k0+16*half .. +16); p points at k0+16*half.
__device__ __forceinline__ bf16x16 ldA(const unsigned short* p) {
  union { bf16x16 v; bf16x8 h[2]; } u;
  u.h[0] = *(const bf16x8*)(p);
  u.h[1] = *(const bf16x8*)(p + 8);
  return u.v;
}
// B-matrix 32x16 bf16 fragment: lane holds column n=lane&15,
// elems = Bt[n][k0+16*half .. +16) from a transposed (NxK) source; p points at k0+16*half.
__device__ __forceinline__ bf16x16 ldB(const unsigned short* p) {
  union { bf16x16 v; bf16x8 h[2]; } u;
  u.h[0] = *(const bf16x8*)(p);
  u.h[1] = *(const bf16x8*)(p + 8);
  return u.v;
}
__device__ __forceinline__ f32x8 wmma_bf16(bf16x16 a, bf16x16 b, f32x8 c) {
  return __builtin_amdgcn_wmma_f32_16x16x32_bf16(false, a, false, b, (short)0, c,
                                                 false, false);
}

// ---------------------------------------------------------------- prep kernels
__global__ void cvt_bf16_kernel(const float* __restrict__ in,
                                unsigned short* __restrict__ out, int n) {
  int i = blockIdx.x * blockDim.x + threadIdx.x;
  if (i < n) out[i] = f2bf(in[i]);
}

__global__ void transpose_bf16_kernel(const float* __restrict__ W,
                                      unsigned short* __restrict__ WT,
                                      int K, int N) {
  int n = blockIdx.x * 16 + (threadIdx.x & 15);
  int k = blockIdx.y * 16 + (threadIdx.x >> 4);
  if (k < K && n < N) WT[(size_t)n * K + k] = f2bf(W[(size_t)k * N + n]);
}

__global__ void posemb_kernel(unsigned short* __restrict__ posb) {
  int i = blockIdx.x * blockDim.x + threadIdx.x;
  if (i >= SEQ * DIM) return;
  int s = i / DIM, d = i % DIM;
  int idx = (d < DIM / 2) ? d : d - DIM / 2;
  // inv_freq = 10000^(-2*idx/DIM) = exp(-idx * 2*ln(1e4)/DIM)
  float inv = __expf(-(float)idx * (2.0f * 9.210340371976184f / (float)DIM));
  float ang = (float)s * inv;
  float val = (d < DIM / 2) ? sinf(ang) : cosf(ang);
  posb[i] = f2bf(val);
}

// ---------------------------------------------------------------- GEMM kernel
// C[M,N] = A[M,K](bf16 row-major) @ Bt[N,K]^T (bf16, stored transposed) + bias
// mode 0: bf16 store row-major          mode 1: store C+u and C+v (Q path)
// mode 2: bf16 store transposed per head (V path: [b][col][s])
// mode 3: fp32 store row-major (final projection)
__global__ void __launch_bounds__(128)
gemm_bf16_kernel(const unsigned short* __restrict__ A,
                 const unsigned short* __restrict__ Bt,
                 const float* __restrict__ bias,
                 int M, int N, int K, int mode, int Srows,
                 unsigned short* __restrict__ out0,
                 unsigned short* __restrict__ out1,
                 float* __restrict__ outf,
                 const float* __restrict__ uvec,
                 const float* __restrict__ vvec) {
  const int lane = threadIdx.x & 31;
  const int wave = threadIdx.x >> 5;
  const int hl   = lane >> 4;
  const int l16  = lane & 15;

  const int m0 = blockIdx.y * 64 + wave * 16;
  const int n0 = blockIdx.x * 64;

  f32x8 acc0 = 0, acc1 = 0, acc2 = 0, acc3 = 0;

  const unsigned short* arow = A + (size_t)(m0 + l16) * K + 16 * hl;
  const unsigned short* b0p  = Bt + (size_t)(n0 +  0 + l16) * K + 16 * hl;
  const unsigned short* b1p  = Bt + (size_t)(n0 + 16 + l16) * K + 16 * hl;
  const unsigned short* b2p  = Bt + (size_t)(n0 + 32 + l16) * K + 16 * hl;
  const unsigned short* b3p  = Bt + (size_t)(n0 + 48 + l16) * K + 16 * hl;

  for (int k0 = 0; k0 < K; k0 += 32) {
    __builtin_prefetch(arow + k0 + 128, 0, 3);   // global_prefetch_b8, near-cache
    __builtin_prefetch(b0p + k0 + 128, 0, 3);
    bf16x16 af = ldA(arow + k0);
    acc0 = wmma_bf16(af, ldB(b0p + k0), acc0);
    acc1 = wmma_bf16(af, ldB(b1p + k0), acc1);
    acc2 = wmma_bf16(af, ldB(b2p + k0), acc2);
    acc3 = wmma_bf16(af, ldB(b3p + k0), acc3);
  }

  f32x8 accs[4] = {acc0, acc1, acc2, acc3};
#pragma unroll
  for (int c = 0; c < 4; ++c) {
    int col = n0 + c * 16 + l16;
    float bz = bias ? bias[col] : 0.0f;
#pragma unroll
    for (int j = 0; j < 8; ++j) {
      int row = m0 + j + 8 * hl;            // C/D layout: vgpr j -> M = j + 8*half
      float v = accs[c][j] + bz;
      if (mode == 0) {
        out0[(size_t)row * N + col] = f2bf(v);
      } else if (mode == 1) {
        out0[(size_t)row * N + col] = f2bf(v + uvec[col]);
        out1[(size_t)row * N + col] = f2bf(v + vvec[col]);
      } else if (mode == 2) {
        int b = row / Srows, s = row % Srows;
        out0[((size_t)b * N + col) * Srows + s] = f2bf(v);
      } else {
        outf[(size_t)row * N + col] = v;
      }
    }
  }
}

// ------------------------------------------------------------ attention kernel
// One workgroup (256 thr = 8 waves, 2 per SIMD32) per (b, h, 16-row s-block).
// LDS: Apre[17][SEQ] bf16 (pos pre-scores rows s0..s0+16), sc[16][SEQ] f32.
__global__ void __launch_bounds__(256)
attn_kernel(const unsigned short* __restrict__ qu,
            const unsigned short* __restrict__ qv,
            const unsigned short* __restrict__ kb,
            const unsigned short* __restrict__ vbT,
            const unsigned short* __restrict__ rb,
            unsigned short* __restrict__ oattn) {
  extern __shared__ char smem[];
  unsigned short* Apre  = (unsigned short*)smem;                 // 17*SEQ bf16
  float*          sc    = (float*)(smem + 17 * SEQ * 2);         // 16*SEQ f32
  unsigned short* probs = (unsigned short*)smem;                 // reuse: 16*SEQ bf16
  __shared__ float red[256];

  const int tid  = threadIdx.x;
  const int lane = tid & 31, wave = tid >> 5;
  const int hl   = lane >> 4, l16 = lane & 15;

  const int nsb  = SEQ / 16;
  const int sblk = blockIdx.x % nsb;
  const int h    = (blockIdx.x / nsb) % NH;
  const int b    = blockIdx.x / (nsb * NH);
  const int s0   = sblk * 16;

  // -------- phase 1: pos pre-scores pre[s,k] = (q_s+v)·r_k, rows s0..s0+16
  bf16x16 avf[2][2];
#pragma unroll
  for (int mt = 0; mt < 2; ++mt) {
    int srow = s0 + mt * 16 + l16;
    if (srow > SEQ - 1) srow = SEQ - 1;                 // clamp (unused rows)
    const unsigned short* p = qv + ((size_t)b * SEQ + srow) * DIM + h * HD + 16 * hl;
    avf[mt][0] = ldA(p);
    avf[mt][1] = ldA(p + 32);
  }
  for (int n0 = wave * 16; n0 < SEQ; n0 += 128) {
    const unsigned short* bp = rb + (size_t)(n0 + l16) * DIM + h * HD + 16 * hl;
    bf16x16 bf0 = ldB(bp), bf1 = ldB(bp + 32);
    f32x8 a0 = 0;
    a0 = wmma_bf16(avf[0][0], bf0, a0);
    a0 = wmma_bf16(avf[0][1], bf1, a0);
#pragma unroll
    for (int j = 0; j < 8; ++j)
      Apre[(j + 8 * hl) * SEQ + n0 + l16] = f2bf(a0[j]);
    f32x8 a1 = 0;
    a1 = wmma_bf16(avf[1][0], bf0, a1);
    a1 = wmma_bf16(avf[1][1], bf1, a1);
    if (hl == 0) Apre[16 * SEQ + n0 + l16] = f2bf(a1[0]);   // row s0+16 only
  }
  __syncthreads();

  // -------- phase 2: content scores + exact relative shift, into sc[16][SEQ]
  bf16x16 auf0, auf1;
  {
    const unsigned short* p = qu + ((size_t)b * SEQ + s0 + l16) * DIM + h * HD + 16 * hl;
    auf0 = ldA(p);
    auf1 = ldA(p + 32);
  }
  for (int t0 = wave * 16; t0 < SEQ; t0 += 128) {
    const unsigned short* bp = kb + ((size_t)b * SEQ + t0 + l16) * DIM + h * HD + 16 * hl;
    f32x8 acc = 0;
    acc = wmma_bf16(auf0, ldB(bp), acc);
    acc = wmma_bf16(auf1, ldB(bp + 32), acc);
    int t = t0 + l16;
#pragma unroll
    for (int j = 0; j < 8; ++j) {
      int si = j + 8 * hl;
      int s  = s0 + si;
      float pterm;
      if (t <= s)          pterm = bf2f(Apre[si * SEQ + (t + SEQ - 1 - s)]);
      else if (t == s + 1) pterm = 0.0f;
      else                 pterm = bf2f(Apre[(si + 1) * SEQ + (t - s - 2)]);
      sc[si * SEQ + t] = (acc[j] + pterm) * 0.125f;   // 1/sqrt(64)
    }
  }
  __syncthreads();

  // -------- phase 3: softmax over full rows (two-pass), probs -> bf16 in LDS
  {
    int r = tid >> 4, c = tid & 15;        // 16 threads per row
    float* row = sc + (size_t)r * SEQ;
    float lmax = -3.0e38f;
    for (int t = c; t < SEQ; t += 16) lmax = fmaxf(lmax, row[t]);
    red[tid] = lmax;
    __syncthreads();
    float rmax = red[r << 4];
#pragma unroll
    for (int i = 1; i < 16; ++i) rmax = fmaxf(rmax, red[(r << 4) + i]);
    __syncthreads();
    float lsum = 0.0f;
    for (int t = c; t < SEQ; t += 16) {
      float e = __expf(row[t] - rmax);
      row[t]  = e;
      lsum   += e;
    }
    red[tid] = lsum;
    __syncthreads();
    float rsum = 0.0f;
#pragma unroll
    for (int i = 0; i < 16; ++i) rsum += red[(r << 4) + i];
    float inv = 1.0f / rsum;
    for (int t = c; t < SEQ; t += 16)
      probs[(size_t)r * SEQ + t] = f2bf(row[t] * inv);
  }
  __syncthreads();

  // -------- phase 4: out[16][64] = probs[16][SEQ] @ V[SEQ][64] (V stored Vt)
  // split-K: 8 waves = 4 n-tiles x 2 k-halves; partials reduced via LDS (sc reuse)
  {
    const int nt = wave & 3;
    const int kh = wave >> 2;
    const int p0 = nt * 16;
    f32x8 acc = 0;
    const unsigned short* bbase =
        vbT + ((size_t)b * DIM + h * HD + p0 + l16) * SEQ + 16 * hl;
    const unsigned short* abase = probs + (size_t)l16 * SEQ + 16 * hl;
    const int kbeg = kh * (SEQ / 2), kend = kbeg + (SEQ / 2);
    for (int kt = kbeg; kt < kend; kt += 32)
      acc = wmma_bf16(ldA(abase + kt), ldB(bbase + kt), acc);
    float* part = sc;                      // score buffer is dead now
    if (kh == 1) {
#pragma unroll
      for (int j = 0; j < 8; ++j) part[nt * 256 + lane * 8 + j] = acc[j];
    }
    __syncthreads();
    if (kh == 0) {
#pragma unroll
      for (int j = 0; j < 8; ++j) {
        float v = acc[j] + part[nt * 256 + lane * 8 + j];
        int s = s0 + j + 8 * hl;
        oattn[((size_t)b * SEQ + s) * DIM + h * HD + p0 + l16] = f2bf(v);
      }
    }
  }
}

// ---------------------------------------------------------------------- launch
extern "C" void kernel_launch(void* const* d_in, const int* in_sizes, int n_in,
                              void* d_out, int out_size, void* d_ws, size_t ws_size,
                              hipStream_t stream) {
  (void)in_sizes; (void)n_in; (void)out_size; (void)ws_size;
  const int M = BATCH * SEQ;                      // 4096 rows
  const float* x  = (const float*)d_in[0];
  const float* Wq = (const float*)d_in[1];
  const float* bq = (const float*)d_in[2];
  const float* Wk = (const float*)d_in[3];
  const float* bk = (const float*)d_in[4];
  const float* Wv = (const float*)d_in[5];
  const float* bv = (const float*)d_in[6];
  const float* Wp = (const float*)d_in[7];
  const float* bp = (const float*)d_in[8];
  const float* Wo = (const float*)d_in[9];
  const float* bo = (const float*)d_in[10];
  const float* uu = (const float*)d_in[11];
  const float* vv = (const float*)d_in[12];

  unsigned short* wsb = (unsigned short*)d_ws;
  size_t off = 0;
  auto alloc = [&](size_t elems) { unsigned short* p = wsb + off; off += elems; return p; };
  unsigned short* xb   = alloc((size_t)M * DIM);
  unsigned short* WqT  = alloc((size_t)DIM * DIM);
  unsigned short* WkT  = alloc((size_t)DIM * DIM);
  unsigned short* WvT  = alloc((size_t)DIM * DIM);
  unsigned short* WpT  = alloc((size_t)DIM * DIM);
  unsigned short* WoT  = alloc((size_t)DIM * DIM);
  unsigned short* posb = alloc((size_t)SEQ * DIM);
  unsigned short* quB  = alloc((size_t)M * DIM);
  unsigned short* qvB  = alloc((size_t)M * DIM);
  unsigned short* kbB  = alloc((size_t)M * DIM);
  unsigned short* vbT  = alloc((size_t)M * DIM);   // [B][D][S] per-head transposed
  unsigned short* rbB  = alloc((size_t)SEQ * DIM);
  unsigned short* oat  = alloc((size_t)M * DIM);

  // prep
  cvt_bf16_kernel<<<(M * DIM + 255) / 256, 256, 0, stream>>>(x, xb, M * DIM);
  dim3 tg(DIM / 16, DIM / 16);
  transpose_bf16_kernel<<<tg, 256, 0, stream>>>(Wq, WqT, DIM, DIM);
  transpose_bf16_kernel<<<tg, 256, 0, stream>>>(Wk, WkT, DIM, DIM);
  transpose_bf16_kernel<<<tg, 256, 0, stream>>>(Wv, WvT, DIM, DIM);
  transpose_bf16_kernel<<<tg, 256, 0, stream>>>(Wp, WpT, DIM, DIM);
  transpose_bf16_kernel<<<tg, 256, 0, stream>>>(Wo, WoT, DIM, DIM);
  posemb_kernel<<<(SEQ * DIM + 255) / 256, 256, 0, stream>>>(posb);

  // projections (bf16 WMMA GEMMs)
  dim3 gA(DIM / 64, M / 64);        // 16 x 64
  dim3 gR(DIM / 64, SEQ / 64);      // 16 x 32
  gemm_bf16_kernel<<<gA, 128, 0, stream>>>(xb, WqT, bq, M, DIM, DIM, 1, SEQ,
                                           quB, qvB, nullptr, uu, vv);
  gemm_bf16_kernel<<<gA, 128, 0, stream>>>(xb, WkT, bk, M, DIM, DIM, 0, SEQ,
                                           kbB, nullptr, nullptr, nullptr, nullptr);
  gemm_bf16_kernel<<<gA, 128, 0, stream>>>(xb, WvT, bv, M, DIM, DIM, 2, SEQ,
                                           vbT, nullptr, nullptr, nullptr, nullptr);
  gemm_bf16_kernel<<<gR, 128, 0, stream>>>(posb, WpT, bp, SEQ, DIM, DIM, 0, SEQ,
                                           rbB, nullptr, nullptr, nullptr, nullptr);

  // fused relative attention (big-LDS path: 17*SEQ bf16 + 16*SEQ f32, 8 waves/WG)
  const size_t smem = (size_t)17 * SEQ * 2 + (size_t)16 * SEQ * 4;  // 200704 B
  attn_kernel<<<BATCH * NH * (SEQ / 16), 256, smem, stream>>>(quB, qvB, kbB, vbT,
                                                              rbB, oat);

  // final projection: fp32 output
  gemm_bf16_kernel<<<gA, 128, 0, stream>>>(oat, WoT, bo, M, DIM, DIM, 3, SEQ,
                                           nullptr, nullptr, (float*)d_out,
                                           nullptr, nullptr);
}
